// RecurrentGCN_2963527435016
// MI455X (gfx1250) — matrix operand order repaired and checked by reading, not compiled, not run on hardware
//
#include <hip/hip_runtime.h>

// RecurrentGCN (K=1) fused kernel for gfx1250 (MI455X, wave32, WMMA).
//
// Reference collapses (H0 == 0, K == 1 -> no diffusion, R unused, edge data dead):
//   z   = x @ (Wz[0,0]+Wz[1,0])[:128,:] + bz
//   h   = x @ (Wh[0,0]+Wh[1,0])[:128,:] + bh
//   out = relu((1-sigmoid(z))*tanh(h)) @ Wlin + blin
// Memory-bound: 51.2 MB of x @ 23.3 TB/s dominates; 3.3 GFLOP is trivial for WMMA.
// One streaming pass over x, weights staged in LDS, fully fused epilogue.

typedef _Float16 half4_t  __attribute__((ext_vector_type(4)));
typedef _Float16 half8_t  __attribute__((ext_vector_type(8)));
typedef _Float16 half16_t __attribute__((ext_vector_type(16)));
typedef float    float4_t __attribute__((ext_vector_type(4)));
typedef float    float8_t __attribute__((ext_vector_type(8)));

#define F_IN   128           // used input features (cols 128..191 of W hit H0==0)
#define HID    64
#define NOUT   128           // fused [Wz_eff | Wh_eff]
#define TM     128           // rows per block (8 waves x 16 rows)
#define XPAD   136           // half stride for x tile rows (bank-conflict pad)
#define WPAD   136           // half stride per weight column

__global__ __launch_bounds__(256)
void rgcn_fused_wmma(const float* __restrict__ x,
                     const float* __restrict__ Wz,    // (2,1,192,64) flat
                     const float* __restrict__ Wh,    // (2,1,192,64) flat
                     const float* __restrict__ bz,
                     const float* __restrict__ bh,
                     const float* __restrict__ Wlin,  // (64,1)
                     const float* __restrict__ blin,  // (1,)
                     float* __restrict__ out,
                     int nrows)
{
    extern __shared__ __align__(16) char smem[];
    _Float16* xs     = (_Float16*)smem;            // [TM][XPAD] f16 x tile
    _Float16* ws     = xs + TM * XPAD;             // [NOUT cols][WPAD] column-major B
    float*    wlin_s = (float*)(ws + NOUT * WPAD); // 64
    float*    bz_s   = wlin_s + HID;               // 64
    float*    bh_s   = bz_s + HID;                 // 64

    const int tid = threadIdx.x;
    const int m0  = blockIdx.x * TM;

    // ---- Stage weights: W_eff[k][j] = W[0,0,k,j] + W[1,0,k,j], k<128.
    // Column-major (per WMMA B fragment: two runs of 8 K per lane, 16 apart).
    // (2,1,192,64): [d,0,k,j] -> d*12288 + k*64 + j. Coalesced over j.
    for (int e = tid; e < F_IN * HID; e += 256) {
        int k = e >> 6, j = e & 63;
        float vz = Wz[k * 64 + j] + Wz[12288 + k * 64 + j];
        float vh = Wh[k * 64 + j] + Wh[12288 + k * 64 + j];
        ws[j * WPAD + k]         = (_Float16)vz;   // cols 0..63  -> Z
        ws[(HID + j) * WPAD + k] = (_Float16)vh;   // cols 64..127-> H
    }
    if (tid < HID) {
        wlin_s[tid] = Wlin[tid];
        bz_s[tid]   = bz[tid];
        bh_s[tid]   = bh[tid];
    }

    // ---- Stage x tile (fp32 -> f16), 16B global loads, 8B LDS stores.
    for (int f = tid; f < TM * (F_IN / 4); f += 256) {
        int row = f >> 5;             // 32 float4 per row
        int c4  = (f & 31) << 2;
        int gm  = m0 + row;
        float4_t v = {0.f, 0.f, 0.f, 0.f};
        if (gm < nrows) v = *(const float4_t*)(x + (size_t)gm * F_IN + c4);
        half4_t h;
        h[0] = (_Float16)v[0]; h[1] = (_Float16)v[1];
        h[2] = (_Float16)v[2]; h[3] = (_Float16)v[3];
        *(half4_t*)(xs + row * XPAD + c4) = h;
    }
    __syncthreads();

    // ---- WMMA: each wave computes 16 rows x 128 cols (8 tiles), K=128 in 4 steps.
    const int wave  = tid >> 5;
    const int lane  = tid & 31;
    const int lhalf = lane & 15;
    const int hi    = lane >> 4;      // 0: lanes 0-15, 1: lanes 16-31

    float8_t acc[8] = {};

    const int arow = 16 * wave + lhalf;
    #pragma unroll
    for (int kk = 0; kk < 4; ++kk) {
        // A 16x32 f16 layout: lane lo holds K kk*32+[0..7],[16..23];
        //                     lane hi holds K kk*32+[8..15],[24..31].
        const _Float16* ap = xs + arow * XPAD + kk * 32 + hi * 8;
        half8_t alo = *(const half8_t*)ap;
        half8_t ahi = *(const half8_t*)(ap + 16);
        half16_t a = __builtin_shufflevector(alo, ahi,
            0, 1, 2, 3, 4, 5, 6, 7, 8, 9, 10, 11, 12, 13, 14, 15);

        #pragma unroll
        for (int t = 0; t < 8; ++t) {
            // B 32x16 f16 layout: column N = 16t + (lane&15);
            // lane lo holds K kk*32+[0..7],[16..23]; lane hi K kk*32+[8..15],[24..31].
            const _Float16* bp = ws + (16 * t + lhalf) * WPAD + kk * 32 + hi * 8;
            half8_t blo = *(const half8_t*)bp;
            half8_t bhi = *(const half8_t*)(bp + 16);
            half16_t b = __builtin_shufflevector(blo, bhi,
                0, 1, 2, 3, 4, 5, 6, 7, 8, 9, 10, 11, 12, 13, 14, 15);

            acc[t] = __builtin_amdgcn_wmma_f32_16x16x32_f16(
                /*neg_a=*/false, a, /*neg_b=*/false, b,
                /*c_mod=*/(short)0, acc[t],
                /*reuse_a=*/false, /*reuse_b=*/false);
        }
    }

    // ---- Fused epilogue. C/D layout: acc[t][r] = C[M = r + 8*hi][N = 16t + lhalf].
    // Tiles 0..3 -> z columns j=16t+lhalf; tiles 4..7 -> h columns (same j).
    float partial[8];
    #pragma unroll
    for (int r = 0; r < 8; ++r) partial[r] = 0.f;

    #pragma unroll
    for (int t = 0; t < 4; ++t) {
        int jj   = 16 * t + lhalf;
        float wl  = wlin_s[jj];
        float bzv = bz_s[jj];
        float bhv = bh_s[jj];
        #pragma unroll
        for (int r = 0; r < 8; ++r) {
            float zv = acc[t][r] + bzv;
            float hv = acc[t + 4][r] + bhv;
            float s  = 1.0f / (1.0f + __expf(-zv));
            float g  = (1.0f - s) * tanhf(hv);
            g = g > 0.f ? g : 0.f;                  // relu
            partial[r] = fmaf(g, wl, partial[r]);
        }
    }

    // 16-lane xor reduction (wave32): every lane in a half gets the row sum.
    #pragma unroll
    for (int r = 0; r < 8; ++r) {
        #pragma unroll
        for (int mask = 1; mask < 16; mask <<= 1)
            partial[r] += __shfl_xor(partial[r], mask, 32);
    }

    if (lhalf == 0) {
        float bl = blin[0];
        int base = m0 + 16 * wave + hi * 8;
        #pragma unroll
        for (int r = 0; r < 8; ++r) {
            int row = base + r;
            if (row < nrows) out[row] = partial[r] + bl;
        }
    }
}

extern "C" void kernel_launch(void* const* d_in, const int* in_sizes, int n_in,
                              void* d_out, int out_size, void* d_ws, size_t ws_size,
                              hipStream_t stream) {
    // Input order: x, edge_index, edge_weight, Wz, bz, Wr, br, Wh, bh, Wlin, blin.
    // edge_index/edge_weight: dead code in the reference (K=1, norms unused).
    // Wr/br: R only multiplies H0 == 0 -> no effect on output.
    const float* x    = (const float*)d_in[0];
    const float* Wz   = (const float*)d_in[3];
    const float* bz   = (const float*)d_in[4];
    const float* Wh   = (const float*)d_in[7];
    const float* bh   = (const float*)d_in[8];
    const float* Wlin = (const float*)d_in[9];
    const float* blin = (const float*)d_in[10];
    float* out = (float*)d_out;

    int nrows  = in_sizes[0] / F_IN;            // 100000
    int blocks = (nrows + TM - 1) / TM;         // 782

    size_t lds = (size_t)(TM * XPAD + NOUT * WPAD) * 2   // f16 x tile + weights
               + (size_t)3 * HID * sizeof(float);        // wlin, bz, bh
    rgcn_fused_wmma<<<blocks, 256, lds, stream>>>(x, Wz, Wh, bz, bh, Wlin, blin,
                                                  out, nrows);
}
